// crossAttention_5265629905535
// MI455X (gfx1250) — matrix operand order repaired and checked by reading, not compiled, not run on hardware
//
#include <hip/hip_runtime.h>

// ---------------------------------------------------------------------------
// CDNA5 (gfx1250) cross-attention pipeline:
//   pack(x,y->NHWC bf16; W->[Cout][3][3][Cin] bf16)
//   conv q,k,v : implicit GEMM, v_wmma_f32_16x16x32_bf16, f32 accum,
//                double-buffered LDS, async global->LDS staging (ASYNCcnt)
//   qk = q@k^T / sqrt(2048)      (per (b,c) 30x30x30 -> VALU, LDS-staged)
//   softmax over channel dim     (wave32 shuffle reductions)
//   qkv = qk@v -> written as NHWC bf16
//   conv o     : same WMMA implicit GEMM -> d_out (f32)
// ---------------------------------------------------------------------------

#if defined(__has_builtin)
#  if __has_builtin(__builtin_amdgcn_global_load_async_to_lds_b128) && \
      __has_builtin(__builtin_amdgcn_s_wait_asynccnt)
#    define HAVE_ASYNC_LDS 1
#  endif
#endif
#ifndef HAVE_ASYNC_LDS
#  define HAVE_ASYNC_LDS 0
#endif

typedef int v4i __attribute__((__vector_size__(16)));

// LDS (address_space(3)) typed pointers: keep staging destinations in AS3 so
// the generic->AS3 addrspacecast (with its null-check cndmask chain) happens
// once at setup instead of per async-load in the hot loop.
typedef __attribute__((address_space(3))) unsigned short lds_us;
typedef __attribute__((address_space(3))) v4i           lds_v4i;
typedef __attribute__((address_space(1))) v4i           glb_v4i;

typedef __attribute__((ext_vector_type(16))) __bf16 v16bf;
typedef __attribute__((ext_vector_type(8)))  float  v8f;

struct B32x2 { uint4 lo; uint4 hi; };   // 32 bytes == one v16bf fragment

__device__ __forceinline__ unsigned short f32_to_bf16_rne(float f) {
  unsigned int u = __builtin_bit_cast(unsigned int, f);
  u += 0x7FFFu + ((u >> 16) & 1u);      // round-to-nearest-even
  return (unsigned short)(u >> 16);
}

// ---------------- pack: NCHW f32 -> NHWC bf16 ------------------------------
__global__ __launch_bounds__(256) void pack_nhwc(
    const float* __restrict__ src, unsigned short* __restrict__ dst,
    int Bb, int C, int H, int W)
{
  long long i = (long long)blockIdx.x * blockDim.x + threadIdx.x;
  long long total = (long long)Bb * C * H * W;
  if (i >= total) return;
  int w = (int)(i % W); long long t = i / W;
  int h = (int)(t % H); t /= H;
  int c = (int)(t % C); int b = (int)(t / C);
  dst[(((long long)b * H + h) * W + w) * C + c] = f32_to_bf16_rne(src[i]);
}

// ------------- pack: W [CO][CI][3][3] f32 -> [CO][3][3][CI] bf16 -----------
__global__ __launch_bounds__(256) void pack_w(
    const float* __restrict__ src, unsigned short* __restrict__ dst,
    int CO, int CI)
{
  long long i = (long long)blockIdx.x * blockDim.x + threadIdx.x;
  long long total = (long long)CO * CI * 9;
  if (i >= total) return;
  int kw = (int)(i % 3); long long t = i / 3;
  int kh = (int)(t % 3); t /= 3;
  int ci = (int)(t % CI); int co = (int)(t / CI);
  dst[((long long)co * 9 + kh * 3 + kw) * CI + ci] = f32_to_bf16_rne(src[i]);
}

// ---- stage one 128x64 A tile chunk (64B/thread) + 64x64 B tile chunk ------
__device__ __forceinline__ void stage_tile(
    lds_us* aDst, const unsigned short* __restrict__ ap,
    lds_us* bDst, const unsigned short* __restrict__ bp,
    bool bvalid)
{
#if HAVE_ASYNC_LDS
  #pragma unroll
  for (int j = 0; j < 4; ++j)
    __builtin_amdgcn_global_load_async_to_lds_b128(
        (glb_v4i*)(ap + j * 8), (lds_v4i*)(aDst + j * 8), 0, 0);
  if (bvalid) {
    #pragma unroll
    for (int j = 0; j < 2; ++j)
      __builtin_amdgcn_global_load_async_to_lds_b128(
          (glb_v4i*)(bp + j * 8), (lds_v4i*)(bDst + j * 8), 0, 0);
  }
#else
  v4i av[4];
  #pragma unroll
  for (int j = 0; j < 4; ++j) av[j] = ((const v4i*)ap)[j];
  #pragma unroll
  for (int j = 0; j < 4; ++j) *(lds_v4i*)(aDst + j * 8) = av[j];
  if (bvalid) {
    v4i b0 = ((const v4i*)bp)[0];
    v4i b1 = ((const v4i*)bp)[1];
    *(lds_v4i*)(bDst)     = b0;
    *(lds_v4i*)(bDst + 8) = b1;
  }
#endif
}

// ---- one 64-deep K step: 8x v_wmma_f32_16x16x32_bf16 per wave -------------
__device__ __forceinline__ void compute_step(
    const unsigned short (&At)[128][64], const unsigned short (&Bt)[64][64],
    v8f (&acc)[2][2], int wm, int wn, int l15, int lh)
{
  #pragma unroll
  for (int ks = 0; ks < 2; ++ks) {
    v16bf af[2], bfv[2];
    #pragma unroll
    for (int sm = 0; sm < 2; ++sm) {
      const int rowA = wm * 32 + sm * 16 + l15;
      B32x2 r;   // lanes0-15: K[0..8)+K[16..24); lanes16-31: +8
      r.lo = *(const uint4*)&At[rowA][ks * 32 + lh * 8];
      r.hi = *(const uint4*)&At[rowA][ks * 32 + lh * 8 + 16];
      af[sm] = __builtin_bit_cast(v16bf, r);
    }
    #pragma unroll
    for (int sn = 0; sn < 2; ++sn) {
      const int col = wn * 32 + sn * 16 + l15;
      B32x2 r;   // lanes0-15: K[0..8)+K[16..24); lanes16-31: +8
      r.lo = *(const uint4*)&Bt[col][ks * 32 + lh * 8];
      r.hi = *(const uint4*)&Bt[col][ks * 32 + lh * 8 + 16];
      bfv[sn] = __builtin_bit_cast(v16bf, r);
    }
    #pragma unroll
    for (int sm = 0; sm < 2; ++sm)
      #pragma unroll
      for (int sn = 0; sn < 2; ++sn)
        acc[sm][sn] = __builtin_amdgcn_wmma_f32_16x16x32_bf16(
            false, af[sm], false, bfv[sn], (short)0, acc[sm][sn], false, false);
  }
}

// ---------------- implicit-GEMM conv (3x3 VALID), bf16 WMMA ----------------
// A : [M][Ktot] bf16 weights (Ktot = 9*Cin, k = (kh*3+kw)*Cin + ci)
// Bx: NHWC bf16 input [Bb][Hin][Hin][Cin]
// Out: [Bb][M][Ho][Ho] f32,  Out += bias
// Block: 256 threads = 8 waves; WG tile 128(M) x 64(N); wave tile 32x32.
// K loop: (kh,kw)-outer / ci-inner -> pure pointer increments in the hot loop.
__global__ __launch_bounds__(256) void conv_gemm_bf16(
    const unsigned short* __restrict__ A,
    const unsigned short* __restrict__ Bx,
    const float* __restrict__ bias,
    float* __restrict__ Out,
    int M, int Cin, int Hin, int Ho, int Bb)
{
  const int Ktot = 9 * Cin;
  const int HoHo = Ho * Ho;
  const int Ntot = Bb * HoHo;

  __shared__ __align__(16) unsigned short Abuf[2][128][64]; // 2 x 16KB
  __shared__ __align__(16) unsigned short Bbuf[2][64][64];  // 2 x 8KB

  const int tid  = threadIdx.x;
  const int lane = tid & 31;
  const int wid  = tid >> 5;
  const int wm   = wid >> 1;          // 0..3 : 32-row group
  const int wn   = wid & 1;           // 0..1 : 32-col group
  const int l15  = lane & 15;
  const int lh   = lane >> 4;

  const int m0 = blockIdx.x * 128;
  const int n0 = blockIdx.y * 64;

  // cooperative A staging: 64B per thread
  const int arow = tid >> 1;              // 0..127
  const int acol = (tid & 1) * 32;        // element offset in 64-elem row
  // cooperative B staging: 32B per thread
  const int bcol    = tid >> 2;           // 0..63
  const int bsegoff = (tid & 3) * 16;     // element offset in 64-elem column

  const int nB = n0 + bcol;
  const bool bvalid = nB < Ntot;
  int bb = 0, hp = 0, wp = 0;
  if (bvalid) { bb = nB / HoHo; int r = nB - bb * HoHo; hp = r / Ho; wp = r - hp * Ho; }
  const unsigned short* Bbase =
      Bx + ((long long)(bb * Hin + hp) * Hin + wp) * Cin + bsegoff;

  // AS3-typed staging destinations (addrspacecast once, 32-bit pointers after)
  lds_us* aDst[2] = { (lds_us*)&Abuf[0][arow][acol], (lds_us*)&Abuf[1][arow][acol] };
  lds_us* bDst[2] = { (lds_us*)&Bbuf[0][bcol][bsegoff], (lds_us*)&Bbuf[1][bcol][bsegoff] };

  // out-of-range columns: zero once; staging never touches them again
  if (!bvalid) {
    v4i z = { 0, 0, 0, 0 };
    *(lds_v4i*)(bDst[0]) = z; *(lds_v4i*)(bDst[0] + 8) = z;
    *(lds_v4i*)(bDst[1]) = z; *(lds_v4i*)(bDst[1] + 8) = z;
  }

  v8f acc[2][2] = {};

  const int inner = Cin >> 6;     // ci steps per (kh,kw)
  const int total = inner * 9;

  // staging cursor (next tile to stage)
  const unsigned short* s_ap = A + (long long)(m0 + arow) * Ktot + acol;
  const unsigned short* s_bp = Bbase;
  int s_kh = 0, s_kw = 0, s_ci = 0;
  auto advance = [&]() {
    s_ap += 64;
    s_ci += 64;
    if (s_ci == Cin) {
      s_ci = 0;
      if (++s_kw == 3) { s_kw = 0; ++s_kh; }
      s_bp = Bbase + ((long long)s_kh * Hin + s_kw) * Cin;
    } else {
      s_bp += 64;
    }
  };

  // prologue: stage tile 0 into buffer 0
  stage_tile(aDst[0], s_ap, bDst[0], s_bp, bvalid);
  advance();

  for (int it = 0; it < total; ++it) {
#if HAVE_ASYNC_LDS
    __builtin_amdgcn_s_wait_asynccnt(0);   // tile `it` landed in LDS
#endif
    __syncthreads();
    if (it + 1 < total) {
      stage_tile(aDst[(it + 1) & 1], s_ap, bDst[(it + 1) & 1], s_bp, bvalid);
      advance();
    }
    compute_step(Abuf[it & 1], Bbuf[it & 1], acc, wm, wn, l15, lh);
  }

  // ---- epilogue: D layout lane&15 = N col, vgpr r + 8*(lane>>4) = M row ----
  #pragma unroll
  for (int sn = 0; sn < 2; ++sn) {
    const int ncol = n0 + wn * 32 + sn * 16 + l15;
    if (ncol >= Ntot) continue;
    const int bb2 = ncol / HoHo;
    int r = ncol - bb2 * HoHo;
    const int h2 = r / Ho;
    const int w2 = r - h2 * Ho;
    #pragma unroll
    for (int sm = 0; sm < 2; ++sm) {
      const int mbase = m0 + wm * 32 + sm * 16 + lh * 8;
      #pragma unroll
      for (int rr = 0; rr < 8; ++rr) {
        const int m = mbase + rr;
        Out[(((long long)bb2 * M + m) * Ho + h2) * Ho + w2] = acc[sm][sn][rr] + bias[m];
      }
    }
  }
}

// ---------------- qk = q @ k^T * scale, per (b,c) 30x30 --------------------
__global__ __launch_bounds__(256) void qk_small_gemm(
    const float* __restrict__ q, const float* __restrict__ k,
    float* __restrict__ qk, float scale)
{
  __shared__ float qs[900];
  __shared__ float ks[900];
  const long long bc = blockIdx.x;
  const float* qp = q + bc * 900;
  const float* kp = k + bc * 900;
  float* op = qk + bc * 900;
  for (int i = threadIdx.x; i < 900; i += 256) { qs[i] = qp[i]; ks[i] = kp[i]; }
  __syncthreads();
  for (int idx = threadIdx.x; idx < 900; idx += 256) {
    const int h = idx / 30;
    const int g = idx - h * 30;
    float s = 0.f;
    #pragma unroll
    for (int w = 0; w < 30; ++w) s = fmaf(qs[h * 30 + w], ks[g * 30 + w], s);
    op[idx] = s * scale;
  }
}

// -------------- softmax over channel dim (1024) at each (b,h,g) ------------
__global__ __launch_bounds__(256) void softmax_channels(float* __restrict__ qk)
{
  const int lane = threadIdx.x & 31;
  const int wid  = threadIdx.x >> 5;
  const int pos  = blockIdx.x * 8 + wid;          // 0..3599
  if (pos >= 3600) return;
  const int b  = pos / 900;
  const int hg = pos - b * 900;
  float* base = qk + (long long)b * 1024 * 900 + hg;

  float vals[32];
  float mx = -3.402823466e38f;
  #pragma unroll
  for (int j = 0; j < 32; ++j) {
    vals[j] = base[(long long)(lane + j * 32) * 900];
    mx = fmaxf(mx, vals[j]);
  }
  #pragma unroll
  for (int off = 16; off > 0; off >>= 1) mx = fmaxf(mx, __shfl_xor(mx, off, 32));
  float sum = 0.f;
  #pragma unroll
  for (int j = 0; j < 32; ++j) { vals[j] = __expf(vals[j] - mx); sum += vals[j]; }
  #pragma unroll
  for (int off = 16; off > 0; off >>= 1) sum += __shfl_xor(sum, off, 32);
  const float inv = 1.f / sum;
  #pragma unroll
  for (int j = 0; j < 32; ++j)
    base[(long long)(lane + j * 32) * 900] = vals[j] * inv;
}

// ---- qkv = qk @ v per (b,c); write directly as NHWC bf16 [B][30][30][1024]
__global__ __launch_bounds__(256) void qkv_small_gemm(
    const float* __restrict__ qk, const float* __restrict__ v,
    unsigned short* __restrict__ qkvT)
{
  __shared__ float as_[900];
  __shared__ float vs[900];
  const int bc = blockIdx.x;
  const int b = bc >> 10;
  const int c = bc & 1023;
  const float* ap = qk + (long long)bc * 900;
  const float* vp = v  + (long long)bc * 900;
  for (int i = threadIdx.x; i < 900; i += 256) { as_[i] = ap[i]; vs[i] = vp[i]; }
  __syncthreads();
  for (int idx = threadIdx.x; idx < 900; idx += 256) {
    const int h = idx / 30;
    const int w = idx - h * 30;
    float s = 0.f;
    #pragma unroll
    for (int g = 0; g < 30; ++g) s = fmaf(as_[h * 30 + g], vs[g * 30 + w], s);
    qkvT[((long long)(b * 30 + h) * 30 + w) * 1024 + c] = f32_to_bf16_rne(s);
  }
}

// ---------------------------------------------------------------------------
extern "C" void kernel_launch(void* const* d_in, const int* in_sizes, int n_in,
                              void* d_out, int out_size, void* d_ws, size_t ws_size,
                              hipStream_t stream)
{
  (void)in_sizes; (void)n_in; (void)out_size; (void)ws_size;
  const float* x  = (const float*)d_in[0];
  const float* y  = (const float*)d_in[1];
  const float* Wq = (const float*)d_in[2];
  const float* bq = (const float*)d_in[3];
  const float* Wk = (const float*)d_in[4];
  const float* bk = (const float*)d_in[5];
  const float* Wv = (const float*)d_in[6];
  const float* bv = (const float*)d_in[7];
  const float* Wo = (const float*)d_in[8];
  const float* bo = (const float*)d_in[9];
  float* out = (float*)d_out;

  char* ws = (char*)d_ws;
  size_t off = 0;
  auto take = [&](size_t bytes) -> char* {
    char* p = ws + off;
    off = (off + bytes + 255) & ~(size_t)255;
    return p;
  };

  const long long nX  = 4LL * 2048 * 32 * 32;   // 8,388,608
  const long long nWq = 1024LL * 2048 * 9;      // 18,874,368
  const long long nWo = 1024LL * 1024 * 9;      //  9,437,184
  const long long nQ  = 4LL * 1024 * 30 * 30;   //  3,686,400

  unsigned short* xT   = (unsigned short*)take((size_t)nX * 2);
  unsigned short* yT   = (unsigned short*)take((size_t)nX * 2);
  unsigned short* WqT  = (unsigned short*)take((size_t)nWq * 2);
  unsigned short* WkT  = (unsigned short*)take((size_t)nWq * 2);
  unsigned short* WvT  = (unsigned short*)take((size_t)nWq * 2);
  unsigned short* WoT  = (unsigned short*)take((size_t)nWo * 2);
  float*          qb   = (float*)take((size_t)nQ * 4);
  float*          kb   = (float*)take((size_t)nQ * 4);
  float*          vb   = (float*)take((size_t)nQ * 4);
  float*          qkb  = (float*)take((size_t)nQ * 4);
  unsigned short* qkvT = (unsigned short*)take((size_t)nQ * 2);

  const int TPB = 256;

  // pack activations to NHWC bf16
  {
    int blocks = (int)((nX + TPB - 1) / TPB);
    pack_nhwc<<<blocks, TPB, 0, stream>>>(x, xT, 4, 2048, 32, 32);
    pack_nhwc<<<blocks, TPB, 0, stream>>>(y, yT, 4, 2048, 32, 32);
  }
  // pack weights to [CO][3][3][CI] bf16
  {
    int blocks = (int)((nWq + TPB - 1) / TPB);
    pack_w<<<blocks, TPB, 0, stream>>>(Wq, WqT, 1024, 2048);
    pack_w<<<blocks, TPB, 0, stream>>>(Wk, WkT, 1024, 2048);
    pack_w<<<blocks, TPB, 0, stream>>>(Wv, WvT, 1024, 2048);
    int blocks_o = (int)((nWo + TPB - 1) / TPB);
    pack_w<<<blocks_o, TPB, 0, stream>>>(Wo, WoT, 1024, 1024);
  }

  // q,k,v convolutions: M=1024, K=18432, N=4*30*30=3600
  {
    dim3 g(1024 / 128, (3600 + 63) / 64);   // (8, 57)
    conv_gemm_bf16<<<g, TPB, 0, stream>>>(WqT, xT, bq, qb, 1024, 2048, 32, 30, 4);
    conv_gemm_bf16<<<g, TPB, 0, stream>>>(WkT, xT, bk, kb, 1024, 2048, 32, 30, 4);
    conv_gemm_bf16<<<g, TPB, 0, stream>>>(WvT, yT, bv, vb, 1024, 2048, 32, 30, 4);
  }

  // qk = q @ k^T / sqrt(2048)
  qk_small_gemm<<<4096, TPB, 0, stream>>>(qb, kb, qkb, 0.02209708691f);

  // softmax over channels
  softmax_channels<<<450, TPB, 0, stream>>>(qkb);

  // qkv = qk @ v -> NHWC bf16
  qkv_small_gemm<<<4096, TPB, 0, stream>>>(qkb, vb, qkvT);

  // output convolution: M=1024, K=9216, N=4*28*28=3136 -> d_out f32
  {
    dim3 g(1024 / 128, (3136 + 63) / 64);   // (8, 49)
    conv_gemm_bf16<<<g, TPB, 0, stream>>>(WoT, qkvT, bo, out, 1024, 1024, 30, 28, 4);
  }
}
